// Wav2Vec2Loss_89601607729734
// MI455X (gfx1250) — matrix operand order repaired and checked by reading, not compiled, not run on hardware
//
#include <hip/hip_runtime.h>
#include <hip/hip_bf16.h>
#include <stdint.h>

// ---------------------------------------------------------------------------
// wav2vec2 contrastive loss, restructured as:
//   1) row norms (f32, exact)
//   2) batched Gram GEMM  G[b] = ctx[b]^T * enc[b]   via v_wmma_f32_16x16x32_f16
//      64x128 block tile, 32x64 wave tile (8 WMMA / k-step), b128 LDS traffic
//   3) exact JAX threefry2x32 negative sampling + stable top-100 selection
//   4) gather sims from Gram, log-softmax loss, atomic sum
// ---------------------------------------------------------------------------

#define BB 4
#define CC 512
#define TT 1024
#define KNEG 100

typedef _Float16 v16h __attribute__((ext_vector_type(16)));
typedef _Float16 v8h  __attribute__((ext_vector_type(8)));
typedef float    v8f  __attribute__((ext_vector_type(8)));

// ---------------------------------------------------------------------------
__global__ void zero_out_kernel(float* out, int n) {
    int i = blockIdx.x * blockDim.x + threadIdx.x;
    if (i < n) out[i] = 0.0f;
}

// ---------------------------------------------------------------------------
// Norms: enc/ctx are [B, C, T]; norm over C for each (b, t).
__global__ void norms_kernel(const float* __restrict__ enc,
                             const float* __restrict__ ctx,
                             float* __restrict__ fN,
                             float* __restrict__ cN) {
    int i = blockIdx.x * blockDim.x + threadIdx.x;   // over B*T
    if (i >= BB * TT) return;
    int b = i >> 10;
    int t = i & (TT - 1);
    const float* pe = enc + (size_t)b * CC * TT + t;
    const float* pc = ctx + (size_t)b * CC * TT + t;
    float se = 0.0f, sc = 0.0f;
    #pragma unroll 4
    for (int c = 0; c < CC; ++c) {
        float a = pe[(size_t)c * TT]; se += a * a;
        float d = pc[(size_t)c * TT]; sc += d * d;
    }
    fN[i] = sqrtf(se);
    cN[i] = sqrtf(sc);
}

// ---------------------------------------------------------------------------
// Gram GEMM: G[b, m, n] = sum_c ctx[b, c, m] * enc[b, c, n]
// Block = 128 threads (4 waves). Block tile M=64 x N=128, K-step 32.
// Wave tile 32x64 = 2x4 WMMA accumulators (8 v_wmma per k-step).
// LDS is fragment-major: tile[row][k], row stride 40 halves (80 B) so every
// 8-half k-group is 16-byte aligned:
//   - staging writes one packed v8h (ds_store_b128) per 8 consecutive k
//   - fragments load as 2x ds_load_b128 per 16x16 operand
#define LDSK 40   // padded k-stride in halves

__global__ void __launch_bounds__(128)
gram_kernel(const float* __restrict__ ctx,
            const float* __restrict__ enc,
            float* __restrict__ gram) {
    const int b      = blockIdx.z;
    const int mBlock = blockIdx.y * 64;
    const int nBlock = blockIdx.x * 128;

    __shared__ __align__(16) _Float16 As[64][LDSK];    // [m][k]
    __shared__ __align__(16) _Float16 Bs[128][LDSK];   // [n][k]

    const int tid  = threadIdx.x;
    const int wave = tid >> 5;
    const int lane = tid & 31;
    const int wm   = (wave >> 1) * 32;   // wave sub-tile row offset (0/32)
    const int wn   = (wave & 1) * 64;    // wave sub-tile col offset (0/64)

    v8f acc[2][4] = {};

    const float* aBase = ctx + (size_t)b * CC * TT;
    const float* bBase = enc + (size_t)b * CC * TT;

    // Fragment addressing (ISA 7.12.2, wave32):
    //  A 16x32 f16: lane l holds row m=l%16; halves h -> k = (h>>3)*16+(h&7)+hiA
    //    with hiA = (l>=16)?8:0  => contiguous groups [hiA..hiA+7], [hiA+16..hiA+23]
    //  B 32x16 f16: lane l holds col n=l%16; halves h -> k = (h>>3)*16+(h&7)+hiA
    //    same k order as A => groups [hiA..hiA+7], [hiA+16..hiA+23]
    const int lsel = lane & 15;
    const int hiA  = (lane >> 4) << 3;
    const int hiB  = (lane >> 4) << 4;

    for (int k0 = 0; k0 < CC; k0 += 32) {
        // ---- Stage A tile: each thread packs 8 consecutive k for one m.
        // 64 m * 4 k-groups = 256 v8h groups -> 2 per thread.
        #pragma unroll
        for (int p = 0; p < 2; ++p) {
            int s  = tid + p * 128;
            int mm = s & 63;
            int kg = (s >> 6) << 3;          // 0,8,16,24
            v8h pk;
            #pragma unroll
            for (int j = 0; j < 8; ++j)
                pk[j] = (_Float16)aBase[(size_t)(k0 + kg + j) * TT + mBlock + mm];
            *(v8h*)&As[mm][kg] = pk;
        }
        // ---- Stage B tile: 128 n * 4 k-groups = 512 groups -> 4 per thread.
        #pragma unroll
        for (int p = 0; p < 4; ++p) {
            int s  = tid + p * 128;
            int mm = s & 127;
            int kg = (s >> 7) << 3;
            v8h pk;
            #pragma unroll
            for (int j = 0; j < 8; ++j)
                pk[j] = (_Float16)bBase[(size_t)(k0 + kg + j) * TT + nBlock + mm];
            *(v8h*)&Bs[mm][kg] = pk;
        }
        // Prefetch next k-tile's cache lines (global_prefetch_b8).
        if (k0 + 32 < CC) {
            __builtin_prefetch(&aBase[(size_t)(k0 + 32 + (tid >> 2)) * TT + mBlock + (tid & 3) * 16], 0, 1);
            __builtin_prefetch(&bBase[(size_t)(k0 + 32 + (tid >> 2)) * TT + nBlock + (tid & 3) * 32], 0, 1);
        }
        __syncthreads();

        v16h afrag[2], bfrag[4];
        #pragma unroll
        for (int i = 0; i < 2; ++i) {
            const int row = wm + i * 16 + lsel;
            union { v16h v; v8h h[2]; } ua;
            ua.h[0] = *(const v8h*)&As[row][hiA];
            ua.h[1] = *(const v8h*)&As[row][hiA + 16];
            afrag[i] = ua.v;
        }
        #pragma unroll
        for (int j = 0; j < 4; ++j) {
            const int col = wn + j * 16 + lsel;
            union { v16h v; v8h h[2]; } ub;
            ub.h[0] = *(const v8h*)&Bs[col][hiA];
            ub.h[1] = *(const v8h*)&Bs[col][hiA + 16];
            bfrag[j] = ub.v;
        }

        #pragma unroll
        for (int i = 0; i < 2; ++i)
            #pragma unroll
            for (int j = 0; j < 4; ++j)
                acc[i][j] = __builtin_amdgcn_wmma_f32_16x16x32_f16(
                    /*neg_a=*/false, afrag[i],
                    /*neg_b=*/false, bfrag[j],
                    /*c_mod=*/(short)0, acc[i][j],
                    /*reuse_a=*/false, /*reuse_b=*/false);
        __syncthreads();
    }

    // Write back. C/D layout: VGPR r, lanes 0-15 -> M=r, lanes 16-31 -> M=8+r;
    // N = lane % 16.
    const int rowOff = (lane >> 4) * 8;
    #pragma unroll
    for (int i = 0; i < 2; ++i)
        #pragma unroll
        for (int j = 0; j < 4; ++j)
            #pragma unroll
            for (int r = 0; r < 8; ++r) {
                int row = mBlock + wm + i * 16 + rowOff + r;
                int col = nBlock + wn + j * 16 + lsel;
                gram[((size_t)b * TT + row) * TT + col] = acc[i][j][r];
            }
}

// ---------------------------------------------------------------------------
// Exact JAX threefry2x32 (20 rounds), key schedule ks2 = k0 ^ k1 ^ 0x1BD11BDA.
__device__ __forceinline__ uint32_t rotl32(uint32_t v, int d) {
    return (v << d) | (v >> (32 - d));
}

__device__ __forceinline__ void threefry2x32(uint32_t k0, uint32_t k1,
                                             uint32_t x0, uint32_t x1,
                                             uint32_t& o0, uint32_t& o1) {
    const uint32_t ks[3] = { k0, k1, k0 ^ k1 ^ 0x1BD11BDAu };
    const int r0[4] = {13, 15, 26, 6};
    const int r1[4] = {17, 29, 16, 24};
    uint32_t x[2] = { x0 + ks[0], x1 + ks[1] };
    #pragma unroll
    for (int blk = 0; blk < 5; ++blk) {
        const int* R = (blk & 1) ? r1 : r0;
        #pragma unroll
        for (int r = 0; r < 4; ++r) {
            x[0] += x[1];
            x[1] = rotl32(x[1], R[r]);
            x[1] ^= x[0];
        }
        x[0] += ks[(blk + 1) % 3];
        x[1] += ks[(blk + 2) % 3] + (uint32_t)(blk + 1);
    }
    o0 = x[0]; o1 = x[1];
}

// One block per (b, t) row: generate 1023 uniforms exactly like
// jax.random.uniform(key(42), (B,T,T-1)), then stable top-100 selection
// (ties -> lower index, matching lax.top_k), remap j -> j + (j >= t).
__global__ void __launch_bounds__(256)
sample_neg_kernel(int* __restrict__ negIdx) {
    const int t = blockIdx.x;
    const int b = blockIdx.y;
    const int n = TT - 1;                      // 1023 candidates

    __shared__ float vals[TT - 1];
    __shared__ float rv[256];
    __shared__ int   ri[256];

    const uint64_t total = (uint64_t)BB * TT * (TT - 1);
    const uint64_t half  = total >> 1;
    const uint64_t row   = ((uint64_t)(b * TT + t)) * (uint64_t)n;

    for (int j = threadIdx.x; j < n; j += blockDim.x) {
        uint64_t g = row + (uint64_t)j;
        uint32_t x0, x1, o0, o1;
        bool first;
        if (g < half) { x0 = (uint32_t)g;          x1 = (uint32_t)(g + half); first = true;  }
        else          { x0 = (uint32_t)(g - half); x1 = (uint32_t)g;          first = false; }
        threefry2x32(0u, 42u, x0, x1, o0, o1);
        uint32_t bits = first ? o0 : o1;
        float u = __uint_as_float((bits >> 9) | 0x3f800000u) - 1.0f;   // [0,1)
        vals[j] = u;
    }
    __syncthreads();

    for (int k = 0; k < KNEG; ++k) {
        float bv = -1.0f;
        int   bi = n;
        for (int j = threadIdx.x; j < n; j += blockDim.x) {
            float v = vals[j];
            if (v > bv || (v == bv && j < bi)) { bv = v; bi = j; }
        }
        rv[threadIdx.x] = bv;
        ri[threadIdx.x] = bi;
        __syncthreads();
        for (int s = 128; s > 0; s >>= 1) {
            if ((int)threadIdx.x < s) {
                float ov = rv[threadIdx.x + s];
                int   oi = ri[threadIdx.x + s];
                if (ov > rv[threadIdx.x] ||
                    (ov == rv[threadIdx.x] && oi < ri[threadIdx.x])) {
                    rv[threadIdx.x] = ov;
                    ri[threadIdx.x] = oi;
                }
            }
            __syncthreads();
        }
        if (threadIdx.x == 0) {
            int sel = ri[0];
            vals[sel] = -2.0f;                         // remove from pool
            negIdx[((b * TT + t) * KNEG) + k] = sel + (sel >= t ? 1 : 0);
        }
        __syncthreads();
    }
}

// ---------------------------------------------------------------------------
// Loss: per (b, t) gather 101 sims from Gram, cosine-normalize, /0.1,
// -log_softmax at index 0, mask, atomic sum.
__global__ void __launch_bounds__(128)
loss_kernel(const float* __restrict__ gram,
            const float* __restrict__ cN,
            const float* __restrict__ fN,
            const int* __restrict__ negIdx,
            const unsigned char* __restrict__ mask,
            float* __restrict__ out) {
    const int t = blockIdx.x;
    const int b = blockIdx.y;
    __shared__ float sims[KNEG + 1];

    const int k = threadIdx.x;
    if (k < KNEG + 1) {
        int idx = (k == 0) ? t : negIdx[(b * TT + t) * KNEG + (k - 1)];
        float dot = gram[((size_t)b * TT + t) * TT + idx];
        float s = dot / (cN[b * TT + t] * fN[b * TT + idx]) * 10.0f;  // /0.1
        sims[k] = s;
    }
    __syncthreads();

    if (threadIdx.x == 0) {
        float m = -INFINITY;
        #pragma unroll 4
        for (int i = 0; i < KNEG + 1; ++i) m = fmaxf(m, sims[i]);
        float sum = 0.0f;
        #pragma unroll 4
        for (int i = 0; i < KNEG + 1; ++i) sum += expf(sims[i] - m);
        float loss = (m + logf(sum)) - sims[0];
        if (mask[b * TT + t]) atomicAdd(out, loss);
    }
}

// ---------------------------------------------------------------------------
extern "C" void kernel_launch(void* const* d_in, const int* in_sizes, int n_in,
                              void* d_out, int out_size, void* d_ws, size_t ws_size,
                              hipStream_t stream) {
    const float*         feat_enc = (const float*)d_in[0];          // [B,C,T]
    const float*         feat_ctx = (const float*)d_in[1];          // [B,C,T]
    const unsigned char* mask     = (const unsigned char*)d_in[2];  // [B,T] bool
    float*               out      = (float*)d_out;

    // Workspace layout
    float* gram   = (float*)d_ws;                                  // B*T*T f32 (16 MB)
    float* cNorm  = gram  + (size_t)BB * TT * TT;                  // B*T
    float* fNorm  = cNorm + (size_t)BB * TT;                       // B*T
    int*   negIdx = (int*)(fNorm + (size_t)BB * TT);               // B*T*KNEG

    zero_out_kernel<<<1, 64, 0, stream>>>(out, out_size);

    norms_kernel<<<(BB * TT + 255) / 256, 256, 0, stream>>>(feat_enc, feat_ctx,
                                                            fNorm, cNorm);

    dim3 gGrid(TT / 128, TT / 64, BB);
    gram_kernel<<<gGrid, 128, 0, stream>>>(feat_ctx, feat_enc, gram);

    dim3 sGrid(TT, BB);
    sample_neg_kernel<<<sGrid, 256, 0, stream>>>(negIdx);

    loss_kernel<<<sGrid, 128, 0, stream>>>(gram, cNorm, fNorm, negIdx, mask, out);
}
